// EthicalAttention_64776696758660
// MI455X (gfx1250) — matrix operand order repaired and checked by reading, not compiled, not run on hardware
//
#include <hip/hip_runtime.h>
#include <hip/hip_bf16.h>
#include <cmath>

// ---------------------------------------------------------------------------
// EthicalAttention on MI455X (gfx1250, wave32, WMMA).
//
// Math note: the reference's "moral bias" adds a per-(b,h,query-row) SCALAR to
// every key column of `scores` before softmax.  softmax(x + c) == softmax(x),
// and masked entries are -inf either way, so moral_context / Wmc / bmc / maw
// have zero effect on both outputs.  They are intentionally not read.
//
// Pipeline (all f16 WMMA 16x16x32, f32 accumulate):
//   1) Qh = (query @ Wq + bq)  -> f16, head-major [B*H][S][64]
//   2) Kh = (key   @ Wk + bk)  -> f16, head-major [B*H][S][64]
//   3) Vt = (value @ Wv + bv)  -> f16, head-major TRANSPOSED [B*H][64][S]
//   4) attn_kernel: per (b,h,16 query rows): scores=Q Kh^T/8 into LDS (128KB),
//      softmax, atomicAdd p/H into attn_mean, attended = P @ V -> f16 [B][S][1024]
//   5) out = attended @ Wo + bo -> f32 d_out
// ---------------------------------------------------------------------------

typedef __attribute__((ext_vector_type(16))) _Float16 v16h;
typedef __attribute__((ext_vector_type(8)))  _Float16 half8;
typedef __attribute__((ext_vector_type(8)))  float    v8f;
typedef __attribute__((ext_vector_type(4)))  float    f4;

union V16 { v16h v; half8 h[2]; };

__device__ inline v8f wmma_f16(v16h a, v16h b, v8f c) {
  // (neg_a, A, neg_b, B, c_mod, C, reuse_a, reuse_b)
  return __builtin_amdgcn_wmma_f32_16x16x32_f16(false, a, false, b, (short)0, c,
                                                false, false);
}

// ---------------------------------------------------------------------------
// Generic 4096x1024x1024 GEMM:  Out = A @ W + bias
//   A: [4096][1024] (f32 if !A_HALF, f16 if A_HALF), row-major
//   W: [1024][1024] f32 row-major
// store modes:
//   0: f32 Out[gm][gn]                              (final projection)
//   1: f16 head-major  Out[((b*16+h)*2048+s)*64+d]  (Q, K)
//   2: f16 head-major transposed Out[((b*16+h)*64+d)*2048+s]  (V)
// Block: 256 thr (8 waves), tile 128x64, BK=32; wave = 32x32 (2x2 WMMA tiles).
// ---------------------------------------------------------------------------
template <bool A_HALF>
__global__ __launch_bounds__(256) void gemm_kernel(
    const void* __restrict__ Av, const float* __restrict__ W,
    const float* __restrict__ bias, void* __restrict__ Out, int mode) {
  constexpr int N = 1024, K = 1024;
  constexpr int TM = 128, TN = 64, TK = 32;
  __shared__ _Float16 lA[TM][TK + 8];  // +8 halves pad, keeps 16B alignment
  __shared__ _Float16 lB[TN][TK + 8];  // B tile stored transposed: lB[n][k]

  const int tid  = threadIdx.x;
  const int wave = tid >> 5, lane = tid & 31;
  const int hi   = lane >> 4, lo16 = lane & 15;
  const int bm = blockIdx.x * TM, bn = blockIdx.y * TN;
  const int wm = (wave & 3) * 32, wn = (wave >> 2) * 32;

  v8f acc[2][2] = {};

  const int arow = tid >> 1, acb = (tid & 1) * 16;   // A: 2 thr/row, 16 ea
  const int bnc  = tid & 63, bkb = (tid >> 6) * 8;   // B: 8 elems/thr

  for (int k0 = 0; k0 < K; k0 += TK) {
    if (A_HALF) {
      const _Float16* A = (const _Float16*)Av;
#pragma unroll
      for (int j = 0; j < 16; ++j)
        lA[arow][acb + j] = A[(size_t)(bm + arow) * K + k0 + acb + j];
      if (k0 + TK < K)  // pull next A tile toward L2/L0 (global_prefetch_b8)
        __builtin_prefetch(&A[(size_t)(bm + arow) * K + k0 + TK + acb], 0, 3);
    } else {
      const float* A = (const float*)Av;
#pragma unroll
      for (int j = 0; j < 16; ++j)
        lA[arow][acb + j] = (_Float16)A[(size_t)(bm + arow) * K + k0 + acb + j];
      if (k0 + TK < K)
        __builtin_prefetch(&A[(size_t)(bm + arow) * K + k0 + TK + acb], 0, 3);
    }
#pragma unroll
    for (int j = 0; j < 8; ++j)
      lB[bnc][bkb + j] = (_Float16)W[(size_t)(k0 + bkb + j) * N + bn + bnc];
    if (k0 + TK < K)
      __builtin_prefetch(&W[(size_t)(k0 + TK + bkb) * N + bn + bnc], 0, 3);
    __syncthreads();

    // Fragment loads per ISA 7.12.2 (wave32 16-bit layouts):
    //   A: lane row m=lane%16; halves = k in [(lane/16)*8, +8) and +16.
    //   B: lane col n=lane%16; halves = k in [(lane/16)*8, +8) and +16.
    V16 af[2], bf[2];
#pragma unroll
    for (int mt = 0; mt < 2; ++mt) {
      const _Float16* p = &lA[wm + mt * 16 + lo16][hi * 8];
      af[mt].h[0] = *(const half8*)p;
      af[mt].h[1] = *(const half8*)(p + 16);
    }
#pragma unroll
    for (int nt = 0; nt < 2; ++nt) {
      const _Float16* p = &lB[wn + nt * 16 + lo16][hi * 8];
      bf[nt].h[0] = *(const half8*)p;
      bf[nt].h[1] = *(const half8*)(p + 16);
    }
#pragma unroll
    for (int mt = 0; mt < 2; ++mt)
#pragma unroll
      for (int nt = 0; nt < 2; ++nt)
        acc[mt][nt] = wmma_f16(af[mt].v, bf[nt].v, acc[mt][nt]);
    __syncthreads();
  }

  // C layout: VGPR r -> M = r + 8*(lane/16), N = lane%16.
#pragma unroll
  for (int mt = 0; mt < 2; ++mt)
#pragma unroll
    for (int nt = 0; nt < 2; ++nt) {
      const int gn = bn + wn + nt * 16 + lo16;
      const float bv = bias[gn];
#pragma unroll
      for (int r = 0; r < 8; ++r) {
        const int gm = bm + wm + mt * 16 + r + hi * 8;
        const float v = acc[mt][nt][r] + bv;
        if (mode == 0) {
          ((float*)Out)[(size_t)gm * N + gn] = v;
        } else {
          const int b_ = gm >> 11, s_ = gm & 2047, h_ = gn >> 6, d_ = gn & 63;
          if (mode == 1)
            ((_Float16*)Out)[(((size_t)(b_ * 16 + h_) * 2048 + s_) << 6) + d_] =
                (_Float16)v;
          else
            ((_Float16*)Out)[(((size_t)(b_ * 16 + h_) * 64 + d_) << 11) + s_] =
                (_Float16)v;
        }
      }
    }
}

// ---------------------------------------------------------------------------
// Fused attention: one block (256 thr / 8 waves) per (b, h, 16 query rows).
// Scores strip (16 x 2048 f32 = 128KB) lives in LDS the whole time.
// ---------------------------------------------------------------------------
__global__ __launch_bounds__(256) void attn_kernel(
    const _Float16* __restrict__ qh, const _Float16* __restrict__ kh,
    const _Float16* __restrict__ vt, const int* __restrict__ mask,
    _Float16* __restrict__ ah, float* __restrict__ attn_mean) {
  constexpr int S = 2048, DK = 64;
  __shared__ float sc[16][S];          // 128 KB (gfx1250: 320 KB/WGP)
  __shared__ float red[16][16];
  __shared__ float rowstat[16];
  __shared__ float redacc[4][16][16];  // cross-wave P@V reduction

  const int tid  = threadIdx.x;
  const int wave = tid >> 5, lane = tid & 31;
  const int hi   = lane >> 4, lo16 = lane & 15;
  const int blk = blockIdx.x;
  const int qt  = blk & 127;           // query tile (S/16)
  const int bh  = blk >> 7;            // 0..31
  const int b   = bh >> 4, h = bh & 15;

  // --- Q fragments (16 rows x dk=64 -> two k-steps of 32), loaded once ---
  const _Float16* qbase = qh + ((size_t)bh * S + qt * 16) * DK;
  V16 qa[2];
#pragma unroll
  for (int f = 0; f < 2; ++f) {
    const _Float16* p = qbase + lo16 * DK + f * 32 + hi * 8;
    qa[f].h[0] = *(const half8*)p;
    qa[f].h[1] = *(const half8*)(p + 16);
  }

  // --- scores = Q K^T / 8, masked, into LDS; key tiles strided over waves ---
  const int* mrow = mask + b * S;
  for (int kt = wave; kt < S / 16; kt += 8) {
    const int keyb = kt * 16;
    const _Float16* kbase = kh + ((size_t)bh * S + keyb) * DK;
    if (kt + 8 < S / 16)  // prefetch this wave's next key tile
      __builtin_prefetch(kbase + (size_t)128 * DK + lo16 * DK, 0, 3);
    V16 kf[2];
#pragma unroll
    for (int f = 0; f < 2; ++f) {
      const _Float16* p = kbase + lo16 * DK + f * 32 + hi * 8;
      kf[f].h[0] = *(const half8*)p;
      kf[f].h[1] = *(const half8*)(p + 16);
    }
    v8f c = {};
    c = wmma_f16(qa[0].v, kf[0].v, c);
    c = wmma_f16(qa[1].v, kf[1].v, c);
    const int key = keyb + lo16;
    const bool valid = mrow[key] != 0;
#pragma unroll
    for (int r = 0; r < 8; ++r)
      sc[r + hi * 8][key] = valid ? c[r] * 0.125f : -INFINITY;
  }
  __syncthreads();

  // --- row-wise softmax: 16 threads per row, 128 elements each, float4 ---
  const int rr = tid >> 4, ch = tid & 15, base = ch * 128;
  f4* srow = (f4*)&sc[rr][base];       // 16B aligned (base % 128 == 0)
  float mx = -INFINITY;
  for (int j = 0; j < 32; ++j) {
    const f4 v = srow[j];              // ds_load_b128
    mx = fmaxf(fmaxf(fmaxf(fmaxf(mx, v[0]), v[1]), v[2]), v[3]);
  }
  red[rr][ch] = mx;
  __syncthreads();
  if (ch == 0) {
    float m2 = -INFINITY;
    for (int j = 0; j < 16; ++j) m2 = fmaxf(m2, red[rr][j]);
    rowstat[rr] = m2;
  }
  __syncthreads();
  const float rm = rowstat[rr];
  float ssum = 0.f;
  for (int j = 0; j < 32; ++j) {
    f4 v = srow[j];
#pragma unroll
    for (int i = 0; i < 4; ++i) {
      v[i] = __expf(v[i] - rm);
      ssum += v[i];
    }
    srow[j] = v;                       // ds_store_b128
  }
  red[rr][ch] = ssum;
  __syncthreads();
  if (ch == 0) {
    float t = 0.f;
    for (int j = 0; j < 16; ++j) t += red[rr][j];
    rowstat[rr] = 1.0f / t;
  }
  __syncthreads();
  const float inv = rowstat[rr];
  float* amrow = attn_mean + ((size_t)b * S + qt * 16 + rr) * S + base;
  for (int j = 0; j < 32; ++j) {
    f4 v = srow[j];
#pragma unroll
    for (int i = 0; i < 4; ++i) v[i] *= inv;
    srow[j] = v;
#pragma unroll
    for (int i = 0; i < 4; ++i)        // global_atomic_add_f32 (L2-resident)
      atomicAdd(&amrow[j * 4 + i], v[i] * (1.0f / 16.0f));
  }
  __syncthreads();

  // --- attended = P(16xS) @ V(Sx64); wave = (n-tile 0..3, K-half 0..1) ---
  const int nt = wave & 3, khf = wave >> 2;
  v8f oc = {};
  const _Float16* vbase = vt + ((size_t)bh * DK + nt * 16 + lo16) * S;
  for (int kc = khf * 32; kc < khf * 32 + 32; ++kc) {
    const int k0 = kc * 32;
    // A frag: probs from LDS, f32 -> f16 at load time
    const float* sp = &sc[lo16][k0 + hi * 8];
    const v8f f0 = *(const v8f*)sp;
    const v8f f1 = *(const v8f*)(sp + 16);
    V16 pa;
#pragma unroll
    for (int i = 0; i < 8; ++i) {
      pa.h[0][i] = (_Float16)f0[i];
      pa.h[1][i] = (_Float16)f1[i];
    }
    // B frag: V^T rows are contiguous in k
    const _Float16* vp = vbase + k0 + hi * 8;
    V16 vb;
    vb.h[0] = *(const half8*)vp;
    vb.h[1] = *(const half8*)(vp + 16);
    oc = wmma_f16(pa.v, vb.v, oc);
  }
  if (khf == 1) {
#pragma unroll
    for (int r = 0; r < 8; ++r) redacc[nt][r + hi * 8][lo16] = oc[r];
  }
  __syncthreads();
  if (khf == 0) {
#pragma unroll
    for (int r = 0; r < 8; ++r) {
      const int m = r + hi * 8;
      const float v = oc[r] + redacc[nt][m][lo16];
      ah[((size_t)b * S + qt * 16 + m) * 1024 + h * 64 + nt * 16 + lo16] =
          (_Float16)v;
    }
  }
}

// ---------------------------------------------------------------------------
extern "C" void kernel_launch(void* const* d_in, const int* in_sizes, int n_in,
                              void* d_out, int out_size, void* d_ws,
                              size_t ws_size, hipStream_t stream) {
  (void)in_sizes; (void)n_in; (void)out_size; (void)ws_size;
  const float* query = (const float*)d_in[0];
  const float* key_  = (const float*)d_in[1];
  const float* value = (const float*)d_in[2];
  // d_in[3] moral_context, d_in[13] Wmc, d_in[14] bmc, d_in[15] maw:
  // provably no-ops (softmax shift invariance) -- unused.
  const int*   mask = (const int*)d_in[4];
  const float* Wq = (const float*)d_in[5];
  const float* bq = (const float*)d_in[6];
  const float* Wk = (const float*)d_in[7];
  const float* bk = (const float*)d_in[8];
  const float* Wv = (const float*)d_in[9];
  const float* bv = (const float*)d_in[10];
  const float* Wo = (const float*)d_in[11];
  const float* bo = (const float*)d_in[12];

  float* out       = (float*)d_out;                 // (2,2048,1024)
  float* attn_mean = out + (size_t)4194304;         // (2,2048,2048)

  _Float16* qh = (_Float16*)d_ws;                   // 8 MB each
  _Float16* kh = qh + (size_t)4194304;
  _Float16* vt = kh + (size_t)4194304;
  _Float16* ah = vt + (size_t)4194304;              // total 32 MB of d_ws

  // attn_mean is accumulated with atomics -> zero it every launch.
  hipMemsetAsync(attn_mean, 0, (size_t)8388608 * sizeof(float), stream);

  const dim3 gg(4096 / 128, 1024 / 64);             // (32, 16) tiles
  gemm_kernel<false><<<gg, 256, 0, stream>>>(query, Wq, bq, qh, 1);
  gemm_kernel<false><<<gg, 256, 0, stream>>>(key_, Wk, bk, kh, 1);
  gemm_kernel<false><<<gg, 256, 0, stream>>>(value, Wv, bv, vt, 2);
  attn_kernel<<<4096, 256, 0, stream>>>(qh, kh, vt, mask, ah, attn_mean);
  gemm_kernel<true><<<gg, 256, 0, stream>>>(ah, Wo, bo, out, 0);
}
